// CircuitGNN_4664334483622
// MI455X (gfx1250) — matrix operand order repaired and checked by reading, not compiled, not run on hardware
//
#include <hip/hip_runtime.h>
#include <hip/hip_bf16.h>

// ---------------------------------------------------------------------------
// CircuitGNN forward for MI455X (gfx1250, wave32).
// Dominant cost: fused QKVS projections as f16 WMMA GEMM (f32 accumulate),
// staged through LDS with CDNA5 async global->LDS copies (ASYNCcnt),
// double-buffered with a peeled (branchless) steady-state K-loop.
// ---------------------------------------------------------------------------

typedef __attribute__((ext_vector_type(16))) _Float16 v16h;
typedef __attribute__((ext_vector_type(8)))  float    v8f;

#define DM   256          // node width per layer
#define QKVS 1024         // 4 * DM fused output width
#define NH   4            // heads
#define HID  64           // per-head dim

// GEMM tiling
#define BM 64
#define BN 128
#define BK 32

// ---------------------------------------------------------------------------
// CDNA5 async global->LDS copy (tracked by ASYNCcnt) + waits
// ---------------------------------------------------------------------------
__device__ __forceinline__ void async_copy_b128(const void* gsrc, void* ldst) {
  unsigned lds = (unsigned)(unsigned long long)(uintptr_t)ldst;  // LDS byte offset
  unsigned long long g = (unsigned long long)(uintptr_t)gsrc;
  asm volatile("global_load_async_to_lds_b128 %0, %1, off"
               :: "v"(lds), "v"(g) : "memory");
}

template <int N>
__device__ __forceinline__ void wait_asynccnt() {
#if __has_builtin(__builtin_amdgcn_s_wait_asynccnt)
  __builtin_amdgcn_s_wait_asynccnt((short)N);
#else
  asm volatile("s_wait_asynccnt %0" :: "n"(N));
#endif
}

// ---------------------------------------------------------------------------
// x (N x 12 f32) -> hA (N x 256 f16), cols 12..31 zeroed (K padded to 32)
// ---------------------------------------------------------------------------
__global__ void k_convert_x(const float* __restrict__ x, _Float16* __restrict__ hA, int N) {
  int idx = blockIdx.x * blockDim.x + threadIdx.x;   // N * 32
  if (idx >= N * 32) return;
  int n = idx >> 5, k = idx & 31;
  hA[(size_t)n * DM + k] = (_Float16)((k < 12) ? x[(size_t)n * 12 + k] : 0.0f);
}

// ---------------------------------------------------------------------------
// Pack [Wq;Wk;Wv;Wskip] (each 256 x din f32) into Bt (1024 x 256 f16,
// out-major / k-minor, zero-padded), and concat biases.
// ---------------------------------------------------------------------------
__global__ void k_pack_weights(const float* __restrict__ Wq, const float* __restrict__ bq,
                               const float* __restrict__ Wk, const float* __restrict__ bk,
                               const float* __restrict__ Wv, const float* __restrict__ bv,
                               const float* __restrict__ Ws, const float* __restrict__ bs,
                               _Float16* __restrict__ Bt, float* __restrict__ biasCat, int din) {
  int idx = blockIdx.x * blockDim.x + threadIdx.x;   // 1024 * 256
  if (idx >= QKVS * DM) return;
  int o = idx >> 8;          // 0..1023 output index
  int k = idx & 255;         // k index
  int mat = o >> 8;          // 0..3 which matrix
  int r   = o & 255;         // row within matrix
  const float* W = (mat == 0) ? Wq : (mat == 1) ? Wk : (mat == 2) ? Wv : Ws;
  float v = (k < din) ? W[(size_t)r * din + k] : 0.0f;
  Bt[(size_t)o * DM + k] = (_Float16)v;
  if (k == 0) {
    const float* bb = (mat == 0) ? bq : (mat == 1) ? bk : (mat == 2) ? bv : bs;
    biasCat[o] = bb[r];
  }
}

// ---------------------------------------------------------------------------
// Fused QKVS GEMM: C[N x 1024] = A[N x K] * Bt^T + bias
//   A : f16 row-major, ld = 256 (only first Kdim cols used)
//   Bt: f16, shape 1024 x 256 (out-major), only first Kdim k-cols nonzero
//   C : f32 row-major, ld = 1024. Layout [q | k | v | skip].
// 8 waves / block; wave computes 32x32 via 2x2 v_wmma_f32_16x16x32_f16.
// Tiles staged via double-buffered global_load_async_to_lds_b128 (ASYNCcnt).
// Fragment layouts per CDNA5 ISA 7.12.2 (wave32).
// ---------------------------------------------------------------------------
__global__ __launch_bounds__(256) void k_qkvs_gemm(
    const _Float16* __restrict__ A, const _Float16* __restrict__ Bt,
    const float* __restrict__ bias, float* __restrict__ C, int Nrows, int Kdim) {
  __shared__ __align__(16) _Float16 As[2][BM][BK + 8];   // 80B row stride
  __shared__ __align__(16) _Float16 Bs[2][BN][BK + 8];

  const int tid  = threadIdx.x;
  const int wave = tid >> 5;
  const int lane = tid & 31;
  const int row0 = blockIdx.x * BM;
  const int col0 = blockIdx.y * BN;

  const int wr = (wave & 1) * 32;    // wave row offset within tile
  const int wc = (wave >> 1) * 32;   // wave col offset within tile
  const int hs = lane >> 4;          // lane half-select
  const int ml = lane & 15;

  // per-thread async-copy chunk assignment (16B each)
  const int rA  = tid >> 2, segA = tid & 3;
  const int grA = min(row0 + rA, Nrows - 1);            // clamp: OOB rows never stored
  const _Float16* gA  = A  + (size_t)grA * DM + segA * 8;
  const int cB0 = tid >> 2;            // chunk 0: cols 0..63 of tile
  const int cB1 = 64 + (tid >> 2);     // chunk 1: cols 64..127 of tile
  const _Float16* gB0 = Bt + (size_t)(col0 + cB0) * DM + segA * 8;
  const _Float16* gB1 = Bt + (size_t)(col0 + cB1) * DM + segA * 8;

  v8f acc[2][2] = {};
  const int nSteps = Kdim / BK;

  // one K-step of WMMA on LDS buffer `buf`
  auto compute = [&](int buf) {
    union V16 { v16h v; unsigned u[8]; };
    V16 afrag[2], bfrag[2];
    #pragma unroll
    for (int t = 0; t < 2; ++t) {
      const _Float16* ap = &As[buf][wr + t * 16 + ml][0];
      const _Float16* bp = &Bs[buf][wc + t * 16 + ml][0];
      #pragma unroll
      for (int j = 0; j < 8; ++j) {
        // A: VGPR j holds K = 2j + hs*8 (+8 more for j>=4)
        int ka = 2 * j + hs * 8 + ((j >= 4) ? 8 : 0);
        afrag[t].u[j] = *(const unsigned*)(ap + ka);
        // B: VGPR j holds K = 2j + hs*8 (+8 more for j>=4)
        int kb = 2 * j + hs * 8 + ((j >= 4) ? 8 : 0);
        bfrag[t].u[j] = *(const unsigned*)(bp + kb);
      }
    }
    #pragma unroll
    for (int i = 0; i < 2; ++i)
      #pragma unroll
      for (int j = 0; j < 2; ++j)
        acc[i][j] = __builtin_amdgcn_wmma_f32_16x16x32_f16(
            /*neg_a=*/false, afrag[i].v, /*neg_b=*/false, bfrag[j].v,
            /*c_mod=*/(short)0, acc[i][j], /*reuse_a=*/false, /*reuse_b=*/false);
  };

  // prologue: stage tile 0 into buffer 0
  async_copy_b128(gA,  &As[0][rA][segA * 8]);
  async_copy_b128(gB0, &Bs[0][cB0][segA * 8]);
  async_copy_b128(gB1, &Bs[0][cB1][segA * 8]);

  // branchless steady state: issue s+1, wait tile s, compute tile s
  int s = 0;
  for (; s < nSteps - 1; ++s) {
    const int buf = s & 1;
    const int k1  = (s + 1) * BK;
    async_copy_b128(gA  + k1, &As[buf ^ 1][rA][segA * 8]);
    async_copy_b128(gB0 + k1, &Bs[buf ^ 1][cB0][segA * 8]);
    async_copy_b128(gB1 + k1, &Bs[buf ^ 1][cB1][segA * 8]);
    wait_asynccnt<3>();   // async ops complete in order -> tile s has landed
    __syncthreads();      // ... for every wave in the block
    compute(buf);
    __syncthreads();      // everyone done reading buf before it is refilled
  }
  // peeled last tile
  wait_asynccnt<0>();
  __syncthreads();
  compute(s & 1);

  // ---- store: C/D layout -> lane holds N = lane&15, VGPR i -> M = 8*hs + i
  #pragma unroll
  for (int i = 0; i < 2; ++i) {
    #pragma unroll
    for (int j = 0; j < 2; ++j) {
      int colb = col0 + wc + j * 16 + ml;
      float bv = bias[colb];
      #pragma unroll
      for (int r = 0; r < 8; ++r) {
        int grow = row0 + wr + i * 16 + hs * 8 + r;
        if (grow < Nrows)
          C[(size_t)grow * QKVS + colb] = acc[i][j][r] + bv;
      }
    }
  }
}

// ---------------------------------------------------------------------------
// Segment-max over float via monotone unsigned key (init key = 0 < key(-inf))
// ---------------------------------------------------------------------------
__device__ __forceinline__ unsigned fkey(float f) {
  unsigned b = __float_as_uint(f);
  return (b & 0x80000000u) ? ~b : (b | 0x80000000u);
}
__device__ __forceinline__ float funkey(unsigned k) {
  if (k == 0u) return -1e30f;
  unsigned b = (k & 0x80000000u) ? (k & 0x7fffffffu) : ~k;
  return __uint_as_float(b);
}

// alpha[e,h] = <q[dst,h,:], k[src,h,:]> / 8 ; atomic segment-max into amax
__global__ void k_edge_logits(const float* __restrict__ qkvs, const int* __restrict__ src,
                              const int* __restrict__ dst, float* __restrict__ alpha,
                              unsigned* __restrict__ amax, int E) {
  int idx = blockIdx.x * blockDim.x + threadIdx.x;   // E * 4
  if (idx >= E * NH) return;
  int e = idx >> 2, h = idx & 3;
  int s = src[e], d = dst[e];
  const float4* qp = (const float4*)(qkvs + (size_t)d * QKVS + h * HID);          // q block
  const float4* kp = (const float4*)(qkvs + (size_t)s * QKVS + DM + h * HID);     // k block
  float acc = 0.f;
  #pragma unroll
  for (int i = 0; i < 16; ++i) {
    float4 q4 = qp[i], k4 = kp[i];
    acc += q4.x * k4.x + q4.y * k4.y + q4.z * k4.z + q4.w * k4.w;
  }
  acc *= 0.125f;                                     // 1/sqrt(64)
  alpha[idx] = acc;
  atomicMax(&amax[(size_t)d * NH + h], fkey(acc));
}

// ex = exp(alpha - amax[dst]); denom[dst] += ex (segment softmax)
__global__ void k_edge_exp(const int* __restrict__ dst, float* __restrict__ alpha,
                           const unsigned* __restrict__ amax, float* __restrict__ denom, int E) {
  int idx = blockIdx.x * blockDim.x + threadIdx.x;
  if (idx >= E * NH) return;
  int e = idx >> 2, h = idx & 3;
  int d = dst[e];
  float ex = __expf(alpha[idx] - funkey(amax[(size_t)d * NH + h]));
  alpha[idx] = ex;
  atomicAdd(&denom[(size_t)d * NH + h], ex);
}

// out[dst, :] += (ex/denom[dst]) * v[src, :]
__global__ __launch_bounds__(256) void k_edge_scatter(
    const float* __restrict__ qkvs, const int* __restrict__ src, const int* __restrict__ dst,
    const float* __restrict__ alpha, const float* __restrict__ denom,
    float* __restrict__ outacc) {
  int e = blockIdx.x;
  int c = threadIdx.x;              // 0..255
  int h = c >> 6;
  int s = src[e], d = dst[e];
  float w = alpha[(size_t)e * NH + h] / fmaxf(denom[(size_t)d * NH + h], 1e-20f);
  atomicAdd(&outacc[(size_t)d * DM + c],
            w * qkvs[(size_t)s * QKVS + 2 * DM + c]);   // v block
}

// h_next = relu(attn_out + skip); write f16 for next GEMM; last layer pools
__global__ __launch_bounds__(256) void k_relu_skip(
    const float* __restrict__ outacc, const float* __restrict__ qkvs,
    _Float16* __restrict__ hA, const int* __restrict__ batch,
    float* __restrict__ poolsum, int last) {
  int n = blockIdx.x;
  int c = threadIdx.x;
  float v = outacc[(size_t)n * DM + c] + qkvs[(size_t)n * QKVS + 3 * DM + c];  // skip block
  v = fmaxf(v, 0.0f);
  hA[(size_t)n * DM + c] = (_Float16)v;
  if (last) atomicAdd(&poolsum[(size_t)batch[n] * DM + c], v);
}

__global__ void k_count_nodes(const int* __restrict__ batch, float* __restrict__ cnt, int N) {
  int n = blockIdx.x * blockDim.x + threadIdx.x;
  if (n < N) atomicAdd(&cnt[batch[n]], 1.0f);
}

// tiny dense layer: out[r,o] = act(b[o] + sum_k in[r,k]*W[o,k])
__global__ void k_small_lin(const float* __restrict__ in, const float* __restrict__ W,
                            const float* __restrict__ b, float* __restrict__ out,
                            int din, int dout, int relu) {
  int r = blockIdx.x, o = threadIdx.x;
  if (o >= dout) return;
  float acc = b[o];
  for (int k = 0; k < din; ++k) acc += in[(size_t)r * din + k] * W[(size_t)o * din + k];
  out[(size_t)r * dout + o] = relu ? fmaxf(acc, 0.0f) : acc;
}

// regin[r] = concat(poolsum[r]/cnt[r]  (256), g[r] (64))
__global__ void k_build_regin(const float* __restrict__ poolsum, const float* __restrict__ cnt,
                              const float* __restrict__ g, float* __restrict__ regin) {
  int r = blockIdx.x, c = threadIdx.x;   // 320 threads
  float v;
  if (c < DM) v = poolsum[(size_t)r * DM + c] / fmaxf(cnt[r], 1.0f);
  else        v = g[(size_t)r * 64 + (c - DM)];
  regin[(size_t)r * 320 + c] = v;
}

// ---------------------------------------------------------------------------
// Launch
// Input order (setup_inputs insertion order, recursive):
//  0:x  1:global_features
//  2..9   conv0 {Wq,bq,Wk,bk,Wv,bv,Wskip,bskip}
//  10..17 conv1, 18..25 conv2
//  26..31 gmlp (W0,b0,W1,b1,W2,b2)
//  32..37 reg  (W0,b0,W1,b1,W2,b2)
//  38:edge_index(2xE)  39:batch(N)
// ---------------------------------------------------------------------------
extern "C" void kernel_launch(void* const* d_in, const int* in_sizes, int n_in,
                              void* d_out, int out_size, void* d_ws, size_t ws_size,
                              hipStream_t stream) {
  const int N = in_sizes[0] / 12;
  const int G = in_sizes[1] / 8;
  const int E = in_sizes[38] / 2;

  const float* x  = (const float*)d_in[0];
  const float* gf = (const float*)d_in[1];
  const int* ei   = (const int*)d_in[38];
  const int* src  = ei;
  const int* dst  = ei + E;
  const int* batch = (const int*)d_in[39];

  // workspace carve-out
  char* base = (char*)d_ws;
  size_t off = 0;
  auto alloc = [&](size_t bytes) -> char* {
    char* p = base + off;
    off = (off + bytes + 255) & ~(size_t)255;
    return p;
  };
  _Float16* hA     = (_Float16*)alloc((size_t)N * DM * 2);
  _Float16* Bt     = (_Float16*)alloc((size_t)QKVS * DM * 2);
  float*    biasC  = (float*)   alloc((size_t)QKVS * 4);
  float*    qkvs   = (float*)   alloc((size_t)N * QKVS * 4);
  float*    alpha  = (float*)   alloc((size_t)E * NH * 4);
  unsigned* amax   = (unsigned*)alloc((size_t)N * NH * 4);
  float*    denom  = (float*)   alloc((size_t)N * NH * 4);
  float*    outacc = (float*)   alloc((size_t)N * DM * 4);
  float*    poolsum= (float*)   alloc((size_t)G * DM * 4);
  float*    cnt    = (float*)   alloc((size_t)G * 4);
  float*    g0     = (float*)   alloc((size_t)G * 64 * 4);
  float*    g1     = (float*)   alloc((size_t)G * 64 * 4);
  float*    regin  = (float*)   alloc((size_t)G * 320 * 4);
  float*    r0     = (float*)   alloc((size_t)G * 128 * 4);
  float*    r1     = (float*)   alloc((size_t)G * 128 * 4);
  (void)ws_size;

  // node input -> f16, K padded to 32
  k_convert_x<<<(N * 32 + 255) / 256, 256, 0, stream>>>(x, hA, N);

  // pooling accumulators (zeroed once per call)
  hipMemsetAsync(poolsum, 0, (size_t)G * DM * 4, stream);
  hipMemsetAsync(cnt, 0, (size_t)G * 4, stream);
  k_count_nodes<<<(N + 255) / 256, 256, 0, stream>>>(batch, cnt, N);

  const int rowBlocks = (N + BM - 1) / BM;
  for (int L = 0; L < 3; ++L) {
    const int pb = 2 + L * 8;
    const int Kdim = (L == 0) ? 32 : DM;
    k_pack_weights<<<(QKVS * DM + 255) / 256, 256, 0, stream>>>(
        (const float*)d_in[pb + 0], (const float*)d_in[pb + 1],
        (const float*)d_in[pb + 2], (const float*)d_in[pb + 3],
        (const float*)d_in[pb + 4], (const float*)d_in[pb + 5],
        (const float*)d_in[pb + 6], (const float*)d_in[pb + 7],
        Bt, biasC, (L == 0) ? 12 : DM);

    dim3 grid(rowBlocks, QKVS / BN, 1);
    k_qkvs_gemm<<<grid, 256, 0, stream>>>(hA, Bt, biasC, qkvs, N, Kdim);

    hipMemsetAsync(amax, 0, (size_t)N * NH * 4, stream);
    hipMemsetAsync(denom, 0, (size_t)N * NH * 4, stream);
    hipMemsetAsync(outacc, 0, (size_t)N * DM * 4, stream);

    k_edge_logits<<<(E * NH + 255) / 256, 256, 0, stream>>>(qkvs, src, dst, alpha, amax, E);
    k_edge_exp<<<(E * NH + 255) / 256, 256, 0, stream>>>(dst, alpha, amax, denom, E);
    k_edge_scatter<<<E, 256, 0, stream>>>(qkvs, src, dst, alpha, denom, outacc);
    k_relu_skip<<<N, 256, 0, stream>>>(outacc, qkvs, hA, batch, poolsum, (L == 2) ? 1 : 0);
  }

  // global MLP branch: 8 -> 64 -> 64 -> 64 (relu each)
  k_small_lin<<<G, 64, 0, stream>>>(gf, (const float*)d_in[26], (const float*)d_in[27], g0, 8, 64, 1);
  k_small_lin<<<G, 64, 0, stream>>>(g0, (const float*)d_in[28], (const float*)d_in[29], g1, 64, 64, 1);
  k_small_lin<<<G, 64, 0, stream>>>(g1, (const float*)d_in[30], (const float*)d_in[31], g0, 64, 64, 1);

  // regressor: concat(mean-pool, g) -> 128 -> 128 -> 1
  k_build_regin<<<G, 320, 0, stream>>>(poolsum, cnt, g0, regin);
  k_small_lin<<<G, 128, 0, stream>>>(regin, (const float*)d_in[32], (const float*)d_in[33], r0, 320, 128, 1);
  k_small_lin<<<G, 128, 0, stream>>>(r0, (const float*)d_in[34], (const float*)d_in[35], r1, 128, 128, 1);
  k_small_lin<<<G, 32, 0, stream>>>(r1, (const float*)d_in[36], (const float*)d_in[37], (float*)d_out, 128, 1, 0);
  (void)out_size;
}
